// DecoderBlock_57492432224478
// MI455X (gfx1250) — compile-verified, matrix operand order checked by reading, not run on hardware
//
#include <hip/hip_runtime.h>
#include <cstddef>

typedef unsigned short u16;
typedef __attribute__((ext_vector_type(16))) __bf16 bf16x16;
typedef __attribute__((ext_vector_type(8)))  float   floatx8;
typedef __attribute__((ext_vector_type(4)))  unsigned int v4u;
typedef __attribute__((ext_vector_type(8)))  int          v8i;
typedef __attribute__((ext_vector_type(4)))  int          v4i;

union Frag16 {
  uint4   u[2];
  bf16x16 v;
};

__device__ __forceinline__ u16 f2bf(float f) {
  unsigned u = __float_as_uint(f);
  unsigned r = u + 0x7FFFu + ((u >> 16) & 1u);   // round-to-nearest-even
  return (u16)(r >> 16);
}

__device__ __forceinline__ floatx8 zero8() {
  floatx8 z;
#pragma unroll
  for (int i = 0; i < 8; ++i) z[i] = 0.f;
  return z;
}

// ---------------------------------------------------------------------------
// TDM: issue a 2D tile load (rows=128, cols=32 bf16) global -> LDS.
// Padded LDS layout: after each 64B row (16 DWORDs), skip 16B (4 DWORDs)
// => row stride 40 bf16 elements, matching __shared__ u16 [128][40].
// D# packing per CDNA5 ISA ch.8 (group0 128b, group1 256b; groups 2/3 zero).
// ---------------------------------------------------------------------------
__device__ __forceinline__ void tdm_load_tile(const u16* g, unsigned lds_byte,
                                              unsigned k_len,
                                              unsigned row_stride_elems) {
  const unsigned long long ga = (unsigned long long)(size_t)g;
  v4u g0;
  g0[0] = 1u;                                   // count=1, user descriptor
  g0[1] = lds_byte;                             // lds_addr
  g0[2] = (unsigned)(ga & 0xFFFFFFFFu);         // global_addr[31:0]
  g0[3] = (unsigned)((ga >> 32) & 0x01FFFFFFu)  // global_addr[56:32]
        | (2u << 30);                           // type=2 ("image")
  const unsigned dim1 = 0x100000u;              // large; tiles never OOB
  v8i g1;
  g1[0] = (int)((1u << 16)      // data_size = 1 (2 bytes)
              | (1u << 20)      // pad_enable
              | (3u << 22)      // pad_interval: 16 DWORDs (one 32-elem row)
              | (3u << 25));    // pad_amount: 4 DWORDs (8 elems)
  g1[1] = (int)(k_len << 16);                         // tensor_dim0 lo16
  g1[2] = (int)((k_len >> 16) | ((dim1 & 0xFFFFu) << 16));
  g1[3] = (int)((dim1 >> 16) | (32u << 16));          // tile_dim0 = 32
  g1[4] = (int)128;                                   // tile_dim1 = 128
  g1[5] = (int)row_stride_elems;                      // tensor_dim0_stride lo32
  g1[6] = 0;                                          // stride hi / dim1_stride
  g1[7] = 0;
  v4i z = {0, 0, 0, 0};
#if __clang_major__ >= 23
  v8i z8 = {0, 0, 0, 0, 0, 0, 0, 0};
  __builtin_amdgcn_tensor_load_to_lds(g0, g1, z, z, z8, 0);
#else
  __builtin_amdgcn_tensor_load_to_lds(g0, g1, z, z, 0);
#endif
}

// ---------------------------------------------------------------------------
// fp32 -> bf16 elementwise conversion (weights / one-time activations)
// ---------------------------------------------------------------------------
__global__ __launch_bounds__(256) void cvt_bf16(const float* __restrict__ in,
                                                u16* __restrict__ out, int n) {
  const int i = (blockIdx.x * 256 + threadIdx.x) * 4;
  if (i < n) {
    const float4 f = *(const float4*)(in + i);
    ushort4 s;
    s.x = f2bf(f.x); s.y = f2bf(f.y); s.z = f2bf(f.z); s.w = f2bf(f.w);
    *(ushort4*)(out + i) = s;
  }
}

// ---------------------------------------------------------------------------
// LayerNorm: one block per row, D=1024, unbiased std (ddof=1), eps on std.
// Emits fp32 (for residuals) and bf16 (GEMM A operand) outputs.
// ---------------------------------------------------------------------------
__global__ __launch_bounds__(256) void ln_kernel(const float* __restrict__ x,
                                                 const float* __restrict__ gamma,
                                                 const float* __restrict__ beta,
                                                 float* __restrict__ y,
                                                 u16* __restrict__ yb) {
  const int row = blockIdx.x;
  const int tid = threadIdx.x;
  const float* xr = x + (size_t)row * 1024;
  float v[4];
  float s = 0.f, s2 = 0.f;
#pragma unroll
  for (int i = 0; i < 4; ++i) {
    v[i] = xr[tid + i * 256];
    s += v[i];
    s2 += v[i] * v[i];
  }
  __shared__ float rs[256];
  __shared__ float rq[256];
  rs[tid] = s; rq[tid] = s2;
  __syncthreads();
  for (int o = 128; o > 0; o >>= 1) {
    if (tid < o) { rs[tid] += rs[tid + o]; rq[tid] += rq[tid + o]; }
    __syncthreads();
  }
  const float mean = rs[0] * (1.f / 1024.f);
  const float var  = (rq[0] - 1024.f * mean * mean) * (1.f / 1023.f);
  const float inv  = 1.f / (sqrtf(fmaxf(var, 0.f)) + 1e-6f);
#pragma unroll
  for (int i = 0; i < 4; ++i) {
    const int c = tid + i * 256;
    const float o = gamma[c] * ((v[i] - mean) * inv) + beta[c];
    y[(size_t)row * 1024 + c] = o;
    yb[(size_t)row * 1024 + c] = f2bf(o);
  }
}

// ---------------------------------------------------------------------------
// WMMA GEMM:  C[m,n] = sum_k A[m,k] * W[n,k]  (+bias, +residual, GELU)
// A and W are bf16. Block tile 128x128, 8 waves (2x4), wave tile 64x32,
// K-step 32. Staging via TDM (tensor_load_to_lds), double-buffered:
// wave 0 prefetches tile i+1 while the WGP computes tile i.
// OMODE: 0 = fp32 row-major, 1 = bf16 row-major, 2 = bf16 [B,H,S,64] heads.
// ---------------------------------------------------------------------------
template <bool GELU_, int OMODE>
__global__ __launch_bounds__(256) void gemm_wmma(
    const u16* __restrict__ A, const u16* __restrict__ W,
    const float* __restrict__ bias, const float* __restrict__ resid,
    float* __restrict__ outF, u16* __restrict__ outB,
    int M, int N, int K) {
  __shared__ u16 As[2][128][40];
  __shared__ u16 Bs[2][128][40];
  const int tid = threadIdx.x, lane = tid & 31, wid = tid >> 5;
  const int wm = wid >> 2, wn = wid & 3;
  const int m0 = blockIdx.y * 128, n0 = blockIdx.x * 128;
  const int ml = lane & 15, k0 = (lane >> 4) * 8;

  const u16* Abase = A + (size_t)m0 * K;
  const u16* Wbase = W + (size_t)n0 * K;

  floatx8 c[4][2];
#pragma unroll
  for (int i = 0; i < 4; ++i)
#pragma unroll
    for (int j = 0; j < 2; ++j) c[i][j] = zero8();

  const int nk = K >> 5;
  if (wid == 0) {
    tdm_load_tile(Abase, (unsigned)(size_t)&As[0][0][0], (unsigned)K, (unsigned)K);
    tdm_load_tile(Wbase, (unsigned)(size_t)&Bs[0][0][0], (unsigned)K, (unsigned)K);
  }

  for (int i = 0; i < nk; ++i) {
    const int p = i & 1;
    if (wid == 0) {
      if (i + 1 < nk) {
        const int kt = (i + 1) * 32;
        tdm_load_tile(Abase + kt, (unsigned)(size_t)&As[p ^ 1][0][0],
                      (unsigned)K, (unsigned)K);
        tdm_load_tile(Wbase + kt, (unsigned)(size_t)&Bs[p ^ 1][0][0],
                      (unsigned)K, (unsigned)K);
        __builtin_amdgcn_s_wait_tensorcnt(2);  // current pair complete
      } else {
        __builtin_amdgcn_s_wait_tensorcnt(0);
      }
    }
    __syncthreads();  // data for buffer p is ready for all waves

    Frag16 a[4], b[2];
#pragma unroll
    for (int mf = 0; mf < 4; ++mf) {
      const u16* pp = &As[p][wm * 64 + mf * 16 + ml][k0];
      a[mf].u[0] = *(const uint4*)pp;
      a[mf].u[1] = *(const uint4*)(pp + 16);
    }
#pragma unroll
    for (int nf = 0; nf < 2; ++nf) {
      const u16* pp = &Bs[p][wn * 32 + nf * 16 + ml][k0];
      b[nf].u[0] = *(const uint4*)pp;
      b[nf].u[1] = *(const uint4*)(pp + 16);
    }
#pragma unroll
    for (int mf = 0; mf < 4; ++mf)
#pragma unroll
      for (int nf = 0; nf < 2; ++nf)
        c[mf][nf] = __builtin_amdgcn_wmma_f32_16x16x32_bf16(
            false, a[mf].v, false, b[nf].v, (short)0, c[mf][nf], false, false);

    __syncthreads();  // all waves done reading buffer p (safe to overwrite)
  }

  // ---- epilogue ----
#pragma unroll
  for (int mf = 0; mf < 4; ++mf) {
#pragma unroll
    for (int nf = 0; nf < 2; ++nf) {
      const int colg = n0 + wn * 32 + nf * 16 + ml;
      const float bb = bias ? bias[colg] : 0.f;
#pragma unroll
      for (int v = 0; v < 8; ++v) {
        const int rowg = m0 + wm * 64 + mf * 16 + ((lane >> 4) << 3) + v;
        float val = c[mf][nf][v] + bb;
        if constexpr (GELU_)
          val = 0.5f * val * (1.f + erff(val * 0.70710678118f));
        if (resid) val += resid[(size_t)rowg * N + colg];
        if constexpr (OMODE == 0) {
          outF[(size_t)rowg * N + colg] = val;
        } else if constexpr (OMODE == 1) {
          outB[(size_t)rowg * N + colg] = f2bf(val);
        } else {  // head-split [B,H,S,64]; M=B*S with S=2048, N=H*64
          const int bI = rowg >> 11, sI = rowg & 2047;
          const int hI = colg >> 6, dI = colg & 63;
          outB[(((size_t)bI * 16 + hI) * 2048 + sI) * 64 + dI] = f2bf(val);
        }
      }
    }
  }
}

// ---------------------------------------------------------------------------
// Flash attention (causal): grid (S/128, B*H); 8 waves x 16 query rows.
// Streaming softmax, WMMA for QK^T and P*V, 32-key tiles staged in LDS.
// ---------------------------------------------------------------------------
__global__ __launch_bounds__(256) void flash_kernel(const u16* __restrict__ Q,
                                                    const u16* __restrict__ K,
                                                    const u16* __restrict__ V,
                                                    u16* __restrict__ ctx) {
  const int bh = blockIdx.y;             // 0..31
  const int bI = bh >> 4, hI = bh & 15;
  const int q0 = blockIdx.x * 128;
  const int tid = threadIdx.x, lane = tid & 31, wid = tid >> 5;
  const int ml = lane & 15, k0 = (lane >> 4) * 8;

  const u16* q_ptr = Q + (size_t)bh * 2048 * 64;
  const u16* k_ptr = K + (size_t)bh * 2048 * 64;
  const u16* v_ptr = V + (size_t)bh * 2048 * 64;

  __shared__ u16 Ks[32][72];      // 32 keys x 64 depth (padded)
  __shared__ u16 Vt[64][40];      // transposed V: depth x 32 keys (padded)
  __shared__ u16 Ps[8][16][40];   // per-wave P tile 16x32 (padded)

  const int qw = q0 + wid * 16;   // this wave's first query row

  // Q fragments (depth 0..31 and 32..63), ISA 16-bit A layout
  Frag16 aq[2];
#pragma unroll
  for (int j = 0; j < 2; ++j) {
    const u16* p = q_ptr + (size_t)(qw + ml) * 64 + j * 32 + k0;
    aq[j].u[0] = *(const uint4*)p;
    aq[j].u[1] = *(const uint4*)(p + 16);
  }

  floatx8 acc[4];
#pragma unroll
  for (int nf = 0; nf < 4; ++nf) acc[nf] = zero8();
  float mi[8], li[8];
#pragma unroll
  for (int v = 0; v < 8; ++v) { mi[v] = -1e30f; li[v] = 0.f; }

  const int kend = q0 + 128;  // causal bound for this block

  for (int kt = 0; kt < kend; kt += 32) {
    __syncthreads();
    // stage K tile: 32x64 bf16, one 16B chunk per thread
    {
      const int row = tid >> 3, cc = (tid & 7) * 8;
      *(uint4*)&Ks[row][cc] =
          *(const uint4*)(k_ptr + (size_t)(kt + row) * 64 + cc);
    }
    // stage V tile transposed (depth-major)
#pragma unroll
    for (int j = 0; j < 8; ++j) {
      const int idx = tid + j * 256;
      const int key = idx & 31, d = idx >> 5;
      Vt[d][key] = v_ptr[(size_t)(kt + key) * 64 + d];
    }
    __syncthreads();

    // S = Q K^T over 32 keys (two 16-wide C tiles), K-dim = depth 64
    floatx8 s[2];
#pragma unroll
    for (int cI = 0; cI < 2; ++cI) {
      floatx8 sc = zero8();
#pragma unroll
      for (int j = 0; j < 2; ++j) {
        Frag16 bk;
        const u16* p = &Ks[cI * 16 + ml][j * 32 + k0];
        bk.u[0] = *(const uint4*)p;
        bk.u[1] = *(const uint4*)(p + 16);
        sc = __builtin_amdgcn_wmma_f32_16x16x32_bf16(
            false, aq[j].v, false, bk.v, (short)0, sc, false, false);
      }
      s[cI] = sc;
    }

    // masked streaming-softmax update (C layout: row = v + 8*(lane>=16))
#pragma unroll
    for (int v = 0; v < 8; ++v) {
      const int m = v + ((lane >> 4) << 3);
      const int qg = qw + m;
      float s0 = s[0][v] * 0.125f;
      float s1 = s[1][v] * 0.125f;
      if (kt + ml > qg) s0 += -1e9f;
      if (kt + 16 + ml > qg) s1 += -1e9f;
      float r = fmaxf(s0, s1);
#pragma unroll
      for (int off = 1; off < 16; off <<= 1) r = fmaxf(r, __shfl_xor(r, off, 32));
      const float mnew = fmaxf(mi[v], r);
      const float alpha = __expf(mi[v] - mnew);
      const float p0 = __expf(s0 - mnew);
      const float p1 = __expf(s1 - mnew);
      float rsum = p0 + p1;
#pragma unroll
      for (int off = 1; off < 16; off <<= 1) rsum += __shfl_xor(rsum, off, 32);
      li[v] = li[v] * alpha + rsum;
      mi[v] = mnew;
#pragma unroll
      for (int nf = 0; nf < 4; ++nf) acc[nf][v] *= alpha;
      Ps[wid][m][ml] = f2bf(p0);
      Ps[wid][m][16 + ml] = f2bf(p1);
    }
    // per-wave LDS round-trip: DS ops are in-order per wave; wait for data
    asm volatile("s_wait_dscnt 0" ::: "memory");

    Frag16 ap;
    {
      const u16* p = &Ps[wid][ml][k0];
      ap.u[0] = *(const uint4*)p;
      ap.u[1] = *(const uint4*)(p + 16);
    }
#pragma unroll
    for (int nf = 0; nf < 4; ++nf) {
      Frag16 bv;
      const u16* p = &Vt[nf * 16 + ml][k0];
      bv.u[0] = *(const uint4*)p;
      bv.u[1] = *(const uint4*)(p + 16);
      acc[nf] = __builtin_amdgcn_wmma_f32_16x16x32_bf16(
          false, ap.v, false, bv.v, (short)0, acc[nf], false, false);
    }
  }

  // write ctx as bf16 in [B,S,H*64] row-major
#pragma unroll
  for (int v = 0; v < 8; ++v) {
    const int m = v + ((lane >> 4) << 3);
    const int qg = qw + m;
    const float inv = 1.f / li[v];
#pragma unroll
    for (int nf = 0; nf < 4; ++nf) {
      ctx[((size_t)bI * 2048 + qg) * 1024 + hI * 64 + nf * 16 + ml] =
          f2bf(acc[nf][v] * inv);
    }
  }
}

// ---------------------------------------------------------------------------
// Host-side orchestration
// ---------------------------------------------------------------------------
extern "C" void kernel_launch(void* const* d_in, const int* in_sizes, int n_in,
                              void* d_out, int out_size, void* d_ws,
                              size_t ws_size, hipStream_t stream) {
  (void)in_sizes; (void)n_in; (void)out_size; (void)ws_size;

  const float* x       = (const float*)d_in[0];
  // d_in[1] = mask (causal, reproduced analytically)
  const float* wq_w    = (const float*)d_in[2];
  const float* wq_b    = (const float*)d_in[3];
  const float* wk_w    = (const float*)d_in[4];
  const float* wk_b    = (const float*)d_in[5];
  const float* wv_w    = (const float*)d_in[6];
  const float* wv_b    = (const float*)d_in[7];
  const float* dense_w = (const float*)d_in[8];
  const float* dense_b = (const float*)d_in[9];
  const float* gamma1  = (const float*)d_in[10];
  const float* beta1   = (const float*)d_in[11];
  const float* gamma2  = (const float*)d_in[12];
  const float* beta2   = (const float*)d_in[13];
  const float* fc_w    = (const float*)d_in[14];
  const float* proj_w  = (const float*)d_in[15];

  constexpr int M = 4096;      // B*S
  constexpr int D = 1024;
  constexpr int F = 4096;      // 4*D

  char* ws = (char*)d_ws;
  size_t off = 0;
  float* xn   = (float*)(ws + off); off += (size_t)M * D * 4;   // xn, later h
  u16*   xnb  = (u16*)  (ws + off); off += (size_t)M * D * 2;
  u16*   qb   = (u16*)  (ws + off); off += (size_t)M * D * 2;
  u16*   kb   = (u16*)  (ws + off); off += (size_t)M * D * 2;
  u16*   vb   = (u16*)  (ws + off); off += (size_t)M * D * 2;
  u16*   ctx  = (u16*)  (ws + off); off += (size_t)M * D * 2;
  float* hn   = (float*)(ws + off); off += (size_t)M * D * 4;
  u16*   hnb  = (u16*)  (ws + off); off += (size_t)M * D * 2;
  u16*   gbuf = (u16*)  (ws + off); off += (size_t)M * F * 2;
  u16*   wqb  = (u16*)  (ws + off); off += (size_t)D * D * 2;
  u16*   wkb  = (u16*)  (ws + off); off += (size_t)D * D * 2;
  u16*   wvb  = (u16*)  (ws + off); off += (size_t)D * D * 2;
  u16*   wdb  = (u16*)  (ws + off); off += (size_t)D * D * 2;
  u16*   wfb  = (u16*)  (ws + off); off += (size_t)F * D * 2;
  u16*   wpb  = (u16*)  (ws + off); off += (size_t)D * F * 2;

  const dim3 blk(256);
  const dim3 gD(D / 128, M / 128);   // (8, 32)
  const dim3 gF(F / 128, M / 128);   // (32, 32)

  // 0) one-time fp32 -> bf16 weight conversion
  cvt_bf16<<<(D * D) / 1024, blk, 0, stream>>>(wq_w, wqb, D * D);
  cvt_bf16<<<(D * D) / 1024, blk, 0, stream>>>(wk_w, wkb, D * D);
  cvt_bf16<<<(D * D) / 1024, blk, 0, stream>>>(wv_w, wvb, D * D);
  cvt_bf16<<<(D * D) / 1024, blk, 0, stream>>>(dense_w, wdb, D * D);
  cvt_bf16<<<(F * D) / 1024, blk, 0, stream>>>(fc_w, wfb, F * D);
  cvt_bf16<<<(D * F) / 1024, blk, 0, stream>>>(proj_w, wpb, D * F);

  // 1) xn = LN1(x)  (fp32 + bf16)
  ln_kernel<<<M, blk, 0, stream>>>(x, gamma1, beta1, xn, xnb);

  // 2) Q/K/V = xn @ W^T + b   -> bf16 [B,H,S,64]
  gemm_wmma<false, 2><<<gD, blk, 0, stream>>>(
      xnb, wqb, wq_b, nullptr, nullptr, qb, M, D, D);
  gemm_wmma<false, 2><<<gD, blk, 0, stream>>>(
      xnb, wkb, wk_b, nullptr, nullptr, kb, M, D, D);
  gemm_wmma<false, 2><<<gD, blk, 0, stream>>>(
      xnb, wvb, wv_b, nullptr, nullptr, vb, M, D, D);

  // 3) causal flash attention -> ctx bf16 [B,S,D]
  flash_kernel<<<dim3(2048 / 128, 32), blk, 0, stream>>>(qb, kb, vb, ctx);

  // 4) h = xn + ctx @ dense_w^T + dense_b   (in place over xn)
  gemm_wmma<false, 0><<<gD, blk, 0, stream>>>(
      ctx, wdb, dense_b, xn, xn, nullptr, M, D, D);

  // 5) hn = LN2(h)  (fp32 + bf16)
  ln_kernel<<<M, blk, 0, stream>>>(xn, gamma2, beta2, hn, hnb);

  // 6) g = gelu(hn @ fc_w^T) -> bf16 [M, 4D]
  gemm_wmma<true, 1><<<gF, blk, 0, stream>>>(
      hnb, wfb, nullptr, nullptr, nullptr, gbuf, M, F, D);

  // 7) out = hn + g @ proj_w^T
  gemm_wmma<false, 0><<<gD, blk, 0, stream>>>(
      gbuf, wpb, nullptr, hn, (float*)d_out, nullptr, M, D, F);
}
